// TransformerBlock_11819749999201
// MI455X (gfx1250) — compile-verified, matrix operand order checked by reading, not run on hardware
//
#include <hip/hip_runtime.h>
#include <hip/hip_bf16.h>

// ---------------------------------------------------------------------------
// Transformer block for MI455X (gfx1250, wave32, WMMA 16x16x32 f16/f32-acc)
// Shapes: B=2, T=2048, EMB=1024, HEADS=16, HDIM=64, FF=4096, M=B*T=4096
// ---------------------------------------------------------------------------

typedef __attribute__((ext_vector_type(16))) _Float16 v16h;
typedef __attribute__((ext_vector_type(8)))  float    v8f;
typedef __attribute__((ext_vector_type(4)))  int      v4i;

// Optional gfx1250 async global->LDS path (builtins present on newer
// toolchains; guarded so we fall back to load+ds_store otherwise).
#if defined(__has_builtin)
#if __has_builtin(__builtin_amdgcn_global_load_async_to_lds_b128) && \
    __has_builtin(__builtin_amdgcn_s_wait_asynccnt)
#define USE_ASYNC_LDS 1
#endif
#endif
#ifndef USE_ASYNC_LDS
#define USE_ASYNC_LDS 0
#endif

typedef __attribute__((address_space(1))) v4i* gvec_ptr;
typedef __attribute__((address_space(3))) v4i* lvec_ptr;

union Frag16 {
    v16h v;
    uint4 q[2];
    _Float16 h[16];
};

__device__ __forceinline__ v8f wmma_f16(v16h a, v16h b, v8f c) {
    // D = A(16x32 f16) * B(32x16 f16) + C(16x16 f32)
    return __builtin_amdgcn_wmma_f32_16x16x32_f16(
        /*neg_a=*/false, a, /*neg_b=*/false, b,
        /*c_mod=*/(short)0, c, /*reuse_a=*/false, /*reuse_b=*/false);
}

// A/B fragment load per ISA 7.12.2: lane l (half = l>>4, idx = l&15) holds
// row (m0+idx); K-groups [k0+half*8,+8) and [k0+16+half*8,+8): two 16B loads.
__device__ __forceinline__ v16h load_frag(const _Float16* __restrict__ base,
                                          int row, int ld, int k0, int lane) {
    const int half = lane >> 4;
    const _Float16* p = base + (size_t)row * ld + k0 + half * 8;
    Frag16 f;
    f.q[0] = *reinterpret_cast<const uint4*>(p);
    f.q[1] = *reinterpret_cast<const uint4*>(p + 16);
    return f.v;
}

__device__ __forceinline__ float gelu_tanh_f(float x) {
    const float c = 0.7978845608028654f; // sqrt(2/pi)
    float x3 = x * x * x;
    return 0.5f * x * (1.0f + tanhf(c * (x + 0.044715f * x3)));
}

// ---------------------------------------------------------------------------
// Weight convert + transpose: W[K,N] fp32 -> Wt[N,K] f16
// ---------------------------------------------------------------------------
__global__ void cvt_transpose_kernel(const float* __restrict__ W,
                                     _Float16* __restrict__ Wt,
                                     int K, int N) {
    size_t id = (size_t)blockIdx.x * blockDim.x + threadIdx.x;
    size_t total = (size_t)K * N;
    if (id >= total) return;
    int n = (int)(id % N);
    int k = (int)(id / N);
    Wt[(size_t)n * K + k] = (_Float16)W[id];
}

// ---------------------------------------------------------------------------
// LayerNorm over last dim (1024), emits f16. One block (256 thr) per row.
// ---------------------------------------------------------------------------
__global__ void ln_kernel(const float* __restrict__ x,
                          const float* __restrict__ sc,
                          const float* __restrict__ sh,
                          _Float16* __restrict__ out) {
    __shared__ float ssum[256];
    __shared__ float ssq[256];
    const int row = blockIdx.x;
    const int tid = threadIdx.x;
    const float* xr = x + (size_t)row * 1024;
    float4 v = reinterpret_cast<const float4*>(xr)[tid];
    ssum[tid] = v.x + v.y + v.z + v.w;
    ssq[tid]  = v.x * v.x + v.y * v.y + v.z * v.z + v.w * v.w;
    __syncthreads();
    for (int s = 128; s > 0; s >>= 1) {
        if (tid < s) { ssum[tid] += ssum[tid + s]; ssq[tid] += ssq[tid + s]; }
        __syncthreads();
    }
    float mean = ssum[0] * (1.0f / 1024.0f);
    float var  = ssq[0] * (1.0f / 1024.0f) - mean * mean;
    float inv  = rsqrtf(var + 1e-5f);
    _Float16* orow = out + (size_t)row * 1024;
    float xv[4] = {v.x, v.y, v.z, v.w};
#pragma unroll
    for (int j = 0; j < 4; ++j) {
        int c = tid * 4 + j;
        orow[c] = (_Float16)(sc[c] * (xv[j] - mean) * inv + sh[c]);
    }
}

// ---------------------------------------------------------------------------
// GEMM: C[M,N] = A[M,K] f16 @ Bt[N,K]^T f16  (+bias) (gelu) (+residual)
// Block = 4 waves -> 64(M) x 64(N) tile. Each wave: 16x64 strip, one shared
// A-fragment reused for 4 WMMAs. B panel (64x32 per K-step) staged to LDS,
// double-buffered, via async global->LDS when available.
// ---------------------------------------------------------------------------
__global__ void gemm_wmma_kernel(const _Float16* __restrict__ A,
                                 const _Float16* __restrict__ Bt,
                                 const float* __restrict__ bias,
                                 const float* __restrict__ residual,
                                 float* __restrict__ Cf,
                                 _Float16* __restrict__ Ch,
                                 int M, int N, int K, int gelu) {
    __shared__ __align__(16) _Float16 Bs[2][64 * 32];

    const int lane = threadIdx.x;            // 0..31
    const int wave = threadIdx.y;            // 0..3
    const int tid  = wave * 32 + lane;       // 0..127
    const int n0 = blockIdx.x * 64;
    const int m0 = blockIdx.y * 64 + wave * 16;
    const int half = lane >> 4;
    const int l15  = lane & 15;

    const _Float16* arow = A + (size_t)(m0 + l15) * K;

    // Stage the 64x32 B panel for K-step k0 into Bs[buf].
    // chunk ch = tid*2+c: row = ch>>2 (0..63), col-group = ch&3 (8 f16 each);
    // LDS byte offset is simply ch*16 (row-major stride 32 f16).
    auto stage_b = [&](int buf, int k0) {
#pragma unroll
        for (int c = 0; c < 2; ++c) {
            int ch  = tid * 2 + c;
            int row = ch >> 2;
            int grp = ch & 3;
            const _Float16* g = Bt + (size_t)(n0 + row) * K + k0 + grp * 8;
            _Float16* l = &Bs[buf][0] + ch * 8;
#if USE_ASYNC_LDS
            __builtin_amdgcn_global_load_async_to_lds_b128(
                (gvec_ptr)g, (lvec_ptr)l, 0, 0);
#else
            *reinterpret_cast<uint4*>(l) = *reinterpret_cast<const uint4*>(g);
#endif
        }
    };

    v8f acc[4] = {{}, {}, {}, {}};

    stage_b(0, 0);
    int buf = 0;
    for (int k0 = 0; k0 < K; k0 += 32) {
        if (k0 + 32 < K) {
            stage_b(buf ^ 1, k0 + 32);
#if USE_ASYNC_LDS
            __builtin_amdgcn_s_wait_asynccnt(2); // prev stage landed
#endif
        } else {
#if USE_ASYNC_LDS
            __builtin_amdgcn_s_wait_asynccnt(0); // final stage landed
#endif
        }
        __syncthreads();

        // A fragment direct from global (per-wave private rows)
        Frag16 fa;
        const _Float16* pa = arow + k0 + half * 8;
        fa.q[0] = *reinterpret_cast<const uint4*>(pa);
        fa.q[1] = *reinterpret_cast<const uint4*>(pa + 16);
        __builtin_prefetch(pa + 128, 0, 0);  // global_prefetch_b8

        // 4 B fragments from LDS, 4 WMMAs reusing fa
#pragma unroll
        for (int s = 0; s < 4; ++s) {
            Frag16 fb;
            const _Float16* pb = &Bs[buf][(s * 16 + l15) * 32 + half * 8];
            fb.q[0] = *reinterpret_cast<const uint4*>(pb);
            fb.q[1] = *reinterpret_cast<const uint4*>(pb + 16);
            acc[s] = wmma_f16(fa.v, fb.v, acc[s]);
        }
        __syncthreads();
        buf ^= 1;
    }

    // epilogue: 4 subtiles of 16x16, C layout row r+8*half, col l15
#pragma unroll
    for (int s = 0; s < 4; ++s) {
        const int n = n0 + s * 16 + l15;
        const float bv = bias ? bias[n] : 0.0f;
#pragma unroll
        for (int r = 0; r < 8; ++r) {
            int m = m0 + r + half * 8;
            float v = acc[s][r] + bv;
            if (gelu) v = gelu_tanh_f(v);
            if (residual) v += residual[(size_t)m * N + n];
            if (Cf) Cf[(size_t)m * N + n] = v;
            if (Ch) Ch[(size_t)m * N + n] = (_Float16)v;
        }
    }
}

// ---------------------------------------------------------------------------
// Pack row-major [M=4096, 1024] f16 into per-head layouts.
// q/k: [B,H,T,D]    vt: [B,H,D,T]
// ---------------------------------------------------------------------------
__global__ void pack_qk_kernel(const _Float16* __restrict__ src,
                               _Float16* __restrict__ dst) {
    size_t id = (size_t)blockIdx.x * blockDim.x + threadIdx.x;
    if (id >= (size_t)4096 * 1024) return;
    int col = (int)(id & 1023);
    int row = (int)(id >> 10);
    int d = col & 63, h = col >> 6;
    int t = row & 2047, b = row >> 11;
    dst[(((size_t)b * 16 + h) * 2048 + t) * 64 + d] = src[id];
}

__global__ void pack_vt_kernel(const _Float16* __restrict__ src,
                               _Float16* __restrict__ dst) {
    size_t id = (size_t)blockIdx.x * blockDim.x + threadIdx.x;
    if (id >= (size_t)4096 * 1024) return;
    int col = (int)(id & 1023);
    int row = (int)(id >> 10);
    int d = col & 63, h = col >> 6;
    int t = row & 2047, b = row >> 11;
    dst[(((size_t)b * 16 + h) * 64 + d) * 2048 + t] = src[id];
}

// ---------------------------------------------------------------------------
// Flash attention: one wave per (b, h, 16-query tile).
// Q,K: [B,H,T,64] f16; Vt: [B,H,64,T] f16; out: [B*T, 1024] f16 (head-merged)
// ---------------------------------------------------------------------------
__global__ void attn_kernel(const _Float16* __restrict__ Q,
                            const _Float16* __restrict__ K,
                            const _Float16* __restrict__ Vt,
                            _Float16* __restrict__ O) {
    const int T = 2048, D = 64;
    const int lane = threadIdx.x;
    const int q0 = blockIdx.x * 16;
    const int h  = blockIdx.y;
    const int b  = blockIdx.z;
    const int half = lane >> 4;
    const int l15  = lane & 15;

    const _Float16* qh = Q  + ((size_t)b * 16 + h) * T * D;
    const _Float16* kh = K  + ((size_t)b * 16 + h) * T * D;
    const _Float16* vh = Vt + ((size_t)b * 16 + h) * D * T;

    __shared__ __align__(16) _Float16 Pbuf[16 * 32];

    // Q fragments for d in [0,32) and [32,64)
    v16h aq0 = load_frag(qh, q0 + l15, D, 0,  lane);
    v16h aq1 = load_frag(qh, q0 + l15, D, 32, lane);

    v8f o0 = {}, o1 = {}, o2 = {}, o3 = {};
    float mi[8], li[8];
#pragma unroll
    for (int r = 0; r < 8; ++r) { mi[r] = -1e30f; li[r] = 0.0f; }

    const float scale = 0.125f;  // 1/sqrt(64)
    const int kend = ((q0 + 16) + 31) & ~31;

    for (int kb = 0; kb < kend; kb += 32) {
        // S = Q(16x64) @ K_blk^T(64x32): two 16-key subtiles, each 2 WMMAs
        v8f s0 = {}, s1 = {};
        {
            v16h bk = load_frag(kh, kb + l15, D, 0, lane);
            s0 = wmma_f16(aq0, bk, s0);
            bk = load_frag(kh, kb + l15, D, 32, lane);
            s0 = wmma_f16(aq1, bk, s0);
        }
        {
            v16h bk = load_frag(kh, kb + 16 + l15, D, 0, lane);
            s1 = wmma_f16(aq0, bk, s1);
            bk = load_frag(kh, kb + 16 + l15, D, 32, lane);
            s1 = wmma_f16(aq1, bk, s1);
        }

        // online softmax: row r+8*half lives on this half-wave, col = l15
#pragma unroll
        for (int r = 0; r < 8; ++r) {
            int m = q0 + r + half * 8;  // global query index
            float a = s0[r] * scale; if (kb + l15 > m)       a = -1e30f;
            float c = s1[r] * scale; if (kb + 16 + l15 > m)  c = -1e30f;
            float vmax = fmaxf(a, c);
            vmax = fmaxf(vmax, __shfl_xor(vmax, 1));
            vmax = fmaxf(vmax, __shfl_xor(vmax, 2));
            vmax = fmaxf(vmax, __shfl_xor(vmax, 4));
            vmax = fmaxf(vmax, __shfl_xor(vmax, 8));
            float newm  = fmaxf(mi[r], vmax);
            float alpha = __expf(mi[r] - newm);
            float p0 = __expf(a - newm);
            float p1 = __expf(c - newm);
            float ps = p0 + p1;
            ps += __shfl_xor(ps, 1);
            ps += __shfl_xor(ps, 2);
            ps += __shfl_xor(ps, 4);
            ps += __shfl_xor(ps, 8);
            li[r] = li[r] * alpha + ps;
            mi[r] = newm;
            o0[r] *= alpha; o1[r] *= alpha; o2[r] *= alpha; o3[r] *= alpha;
            int row = r + half * 8;
            Pbuf[row * 32 + l15]      = (_Float16)p0;
            Pbuf[row * 32 + 16 + l15] = (_Float16)p1;
        }
        __syncthreads();

        // reload P(16x32) as an A fragment from LDS
        Frag16 fp;
        const _Float16* pp = &Pbuf[l15 * 32 + half * 8];
        fp.q[0] = *reinterpret_cast<const uint4*>(pp);
        fp.q[1] = *reinterpret_cast<const uint4*>(pp + 16);
        __syncthreads();

        // O(16x64) += P(16x32) @ V_blk(32x16) per 16-wide d-chunk
        v16h bv;
        bv = load_frag(vh, 0  + l15, T, kb, lane);  o0 = wmma_f16(fp.v, bv, o0);
        bv = load_frag(vh, 16 + l15, T, kb, lane);  o1 = wmma_f16(fp.v, bv, o1);
        bv = load_frag(vh, 32 + l15, T, kb, lane);  o2 = wmma_f16(fp.v, bv, o2);
        bv = load_frag(vh, 48 + l15, T, kb, lane);  o3 = wmma_f16(fp.v, bv, o3);
    }

    // finalize: divide by row sums, write head-merged [B*T, 1024]
#pragma unroll
    for (int r = 0; r < 8; ++r) {
        int t = q0 + r + half * 8;
        float invl = 1.0f / li[r];
        size_t base = ((size_t)b * T + t) * 1024 + (size_t)h * 64;
        O[base + 0  + l15] = (_Float16)(o0[r] * invl);
        O[base + 16 + l15] = (_Float16)(o1[r] * invl);
        O[base + 32 + l15] = (_Float16)(o2[r] * invl);
        O[base + 48 + l15] = (_Float16)(o3[r] * invl);
    }
}

// ---------------------------------------------------------------------------
// Host launch
// ---------------------------------------------------------------------------
extern "C" void kernel_launch(void* const* d_in, const int* in_sizes, int n_in,
                              void* d_out, int out_size, void* d_ws, size_t ws_size,
                              hipStream_t stream) {
    const float* x    = (const float*)d_in[0];
    const float* Wq   = (const float*)d_in[1];
    const float* Wk   = (const float*)d_in[2];
    const float* Wv   = (const float*)d_in[3];
    const float* Wo   = (const float*)d_in[4];
    const float* bo   = (const float*)d_in[5];
    const float* W1   = (const float*)d_in[6];
    const float* b1   = (const float*)d_in[7];
    const float* W2   = (const float*)d_in[8];
    const float* b2   = (const float*)d_in[9];
    const float* ln1s = (const float*)d_in[10];
    const float* ln1b = (const float*)d_in[11];
    const float* ln2s = (const float*)d_in[12];
    const float* ln2b = (const float*)d_in[13];

    const int M = 4096, E = 1024, FFD = 4096;
    char* ws = (char*)d_ws;
    const size_t MB = 1u << 20;
    _Float16* WqT   = (_Float16*)(ws + 0   * MB);  // 2MB
    _Float16* WkT   = (_Float16*)(ws + 2   * MB);  // 2MB
    _Float16* WvT   = (_Float16*)(ws + 4   * MB);  // 2MB
    _Float16* WoT   = (_Float16*)(ws + 6   * MB);  // 2MB
    _Float16* W1T   = (_Float16*)(ws + 8   * MB);  // 8MB
    _Float16* W2T   = (_Float16*)(ws + 16  * MB);  // 8MB
    _Float16* h1    = (_Float16*)(ws + 24  * MB);  // 8MB
    _Float16* qtmp  = (_Float16*)(ws + 32  * MB);  // 8MB
    _Float16* ktmp  = (_Float16*)(ws + 40  * MB);  // 8MB
    _Float16* vtmp  = (_Float16*)(ws + 48  * MB);  // 8MB
    _Float16* qbuf  = (_Float16*)(ws + 56  * MB);  // 8MB
    _Float16* kbuf  = (_Float16*)(ws + 64  * MB);  // 8MB
    _Float16* vtbuf = (_Float16*)(ws + 72  * MB);  // 8MB
    _Float16* attO  = (_Float16*)(ws + 80  * MB);  // 8MB
    float*    x1    = (float*)   (ws + 88  * MB);  // 16MB
    _Float16* h2    = (_Float16*)(ws + 104 * MB);  // 8MB
    _Float16* ff1   = (_Float16*)(ws + 112 * MB);  // 32MB

    // weight conversion + transpose
    {
        int blk = 256;
        int gEE = (E * E + blk - 1) / blk;
        int gEF = (E * FFD + blk - 1) / blk;
        cvt_transpose_kernel<<<gEE, blk, 0, stream>>>(Wq, WqT, E, E);
        cvt_transpose_kernel<<<gEE, blk, 0, stream>>>(Wk, WkT, E, E);
        cvt_transpose_kernel<<<gEE, blk, 0, stream>>>(Wv, WvT, E, E);
        cvt_transpose_kernel<<<gEE, blk, 0, stream>>>(Wo, WoT, E, E);
        cvt_transpose_kernel<<<gEF, blk, 0, stream>>>(W1, W1T, E, FFD);
        cvt_transpose_kernel<<<gEF, blk, 0, stream>>>(W2, W2T, FFD, E);
    }

    dim3 gblk(32, 4);

    // LN1 -> h1 (f16)
    ln_kernel<<<M, 256, 0, stream>>>(x, ln1s, ln1b, h1);

    // QKV projections (f16 out)
    gemm_wmma_kernel<<<dim3(E / 64, M / 64), gblk, 0, stream>>>(
        h1, WqT, nullptr, nullptr, nullptr, qtmp, M, E, E, 0);
    gemm_wmma_kernel<<<dim3(E / 64, M / 64), gblk, 0, stream>>>(
        h1, WkT, nullptr, nullptr, nullptr, ktmp, M, E, E, 0);
    gemm_wmma_kernel<<<dim3(E / 64, M / 64), gblk, 0, stream>>>(
        h1, WvT, nullptr, nullptr, nullptr, vtmp, M, E, E, 0);

    // repack into per-head layouts
    {
        int blk = 256;
        int g = (int)(((size_t)M * E + blk - 1) / blk);
        pack_qk_kernel<<<g, blk, 0, stream>>>(qtmp, qbuf);
        pack_qk_kernel<<<g, blk, 0, stream>>>(ktmp, kbuf);
        pack_vt_kernel<<<g, blk, 0, stream>>>(vtmp, vtbuf);
    }

    // flash attention
    attn_kernel<<<dim3(2048 / 16, 16, 2), 32, 0, stream>>>(qbuf, kbuf, vtbuf, attO);

    // out projection + residual -> x1 (fp32)
    gemm_wmma_kernel<<<dim3(E / 64, M / 64), gblk, 0, stream>>>(
        attO, WoT, bo, x, x1, nullptr, M, E, E, 0);

    // LN2 -> h2 (f16)
    ln_kernel<<<M, 256, 0, stream>>>(x1, ln2s, ln2b, h2);

    // FFN up + GELU (f16 out)
    gemm_wmma_kernel<<<dim3(FFD / 64, M / 64), gblk, 0, stream>>>(
        h2, W1T, b1, nullptr, nullptr, ff1, M, FFD, E, 1);

    // FFN down + residual -> d_out (fp32)
    gemm_wmma_kernel<<<dim3(E / 64, M / 64), gblk, 0, stream>>>(
        ff1, W2T, b2, x1, (float*)d_out, nullptr, M, E, FFD, 0);
}
